// RelativePositionEncoder_7069516169780
// MI455X (gfx1250) — matrix operand order repaired and checked by reading, not compiled, not run on hardware
//
#include <hip/hip_runtime.h>
#include <hip/hip_bf16.h>
#include <math.h>

// ---------------------------------------------------------------------------
// Problem constants (from reference)
// ---------------------------------------------------------------------------
#define BB   8
#define NN   512
#define CSD  256
#define HH   8
#define CHD  32
#define KK   64
#define BINS 129          // 2*K+1
#define CTD  1024
#define OUTD 2
#define CATK 1288         // (BINS+CH)*H
#define CATP 1312         // CATK padded to multiple of 32
#define ROWS (BB*NN)      // 4096

typedef __attribute__((ext_vector_type(16))) unsigned short vu16x16;
typedef __attribute__((ext_vector_type(8)))  unsigned short vu16x8;
typedef __attribute__((ext_vector_type(16))) __bf16        v16bf;
typedef __attribute__((ext_vector_type(8)))  float         v8f;

// ---------------------------------------------------------------------------
// bf16 helpers (round-to-nearest-even)
// ---------------------------------------------------------------------------
__device__ __forceinline__ unsigned short f2bf(float f) {
  union { float f; unsigned u; } c; c.f = f;
  unsigned u = c.u;
  unsigned r = u + 0x7FFFu + ((u >> 16) & 1u);
  return (unsigned short)(r >> 16);
}

// ---------------------------------------------------------------------------
// CDNA5 WMMA bf16 fragments as contiguous vector loads (ISA 7.12.2, wave32).
// A (16x32, row-major, lda elems): lane m=l&15, kb=(l&16)?8:0;
//   holds K = kb+0..7 (16B run) and kb+16..23 (16B run).
// B (32x16) from TRANSPOSED storage BT[n][k]: lane n=l&15, kb=(l&16)?8:0;
//   holds K = kb+0..7 (16B run) and kb+16..23 (16B run).
// Requires lda/ldb multiples of 8 elems, k0 multiple of 32, 16B-aligned base.
// ---------------------------------------------------------------------------
__device__ __forceinline__ vu16x16 load_frag_a(const unsigned short* base,
                                               long lda, int m0, int k0) {
  int l = threadIdx.x & 31;
  const unsigned short* p =
      base + (long)(m0 + (l & 15)) * lda + k0 + ((l & 16) ? 8 : 0);
  vu16x8 lo = *(const vu16x8*)p;
  vu16x8 hi = *(const vu16x8*)(p + 16);
  vu16x16 r;
#pragma unroll
  for (int i = 0; i < 8; ++i) { r[i] = lo[i]; r[8 + i] = hi[i]; }
  return r;
}

__device__ __forceinline__ vu16x16 load_frag_b(const unsigned short* baseT,
                                               long ldb, int n0, int k0) {
  int l = threadIdx.x & 31;
  const unsigned short* p =
      baseT + (long)(n0 + (l & 15)) * ldb + k0 + ((l & 16) ? 8 : 0);
  vu16x8 lo = *(const vu16x8*)p;
  vu16x8 hi = *(const vu16x8*)(p + 16);
  vu16x16 r;
#pragma unroll
  for (int i = 0; i < 8; ++i) { r[i] = lo[i]; r[8 + i] = hi[i]; }
  return r;
}

__device__ __forceinline__ v8f wmma_bf16(vu16x16 a, vu16x16 b, v8f c) {
  return __builtin_amdgcn_wmma_f32_16x16x32_bf16(
      false, __builtin_bit_cast(v16bf, a),
      false, __builtin_bit_cast(v16bf, b),
      (short)0, c, false, false);
}

// ---------------------------------------------------------------------------
// f32 -> bf16 conversion (plain, and transposing for B operands)
// ---------------------------------------------------------------------------
__global__ void cvt_bf16_kernel(const float* __restrict__ in,
                                unsigned short* __restrict__ out, long n) {
  long i = (long)blockIdx.x * blockDim.x + threadIdx.x;
  if (i < n) out[i] = f2bf(in[i]);
}

// in: [K][N] f32 row-major  ->  out: [N][ldo] bf16 (transposed, padded ld)
__global__ void cvt_bf16_t_kernel(const float* __restrict__ in,
                                  unsigned short* __restrict__ out,
                                  int K, int N, int ldo) {
  long i = (long)blockIdx.x * blockDim.x + threadIdx.x;
  if (i < (long)K * N) {
    int k = (int)(i / N), n = (int)(i % N);
    out[(long)n * ldo + k] = f2bf(in[i]);
  }
}

// zero the K-padding: cat cols 1288..1311 (4096 rows), WoT cols 1288..1311
// (256 rows)
__global__ void zero_pads_kernel(unsigned short* __restrict__ cat,
                                 unsigned short* __restrict__ WoT) {
  int i = blockIdx.x * blockDim.x + threadIdx.x;
  if (i < ROWS * (CATP - CATK)) {
    int r = i / (CATP - CATK), c = CATK + i % (CATP - CATK);
    cat[(long)r * CATP + c] = 0;
  }
  if (i < CSD * (CATP - CATK)) {
    int r = i / (CATP - CATK), c = CATK + i % (CATP - CATK);
    WoT[(long)r * CATP + c] = 0;
  }
}

// ---------------------------------------------------------------------------
// bf16 WMMA GEMM: C[M,N] = A[M,K] @ B[K,N], B given TRANSPOSED as BT[N][K].
// block = 128 threads (4 waves); tile = 16 (M) x 64 (N); K step 32.
// No LDS, no barriers: fragments are direct 128-bit vector loads (L2 resident).
// omode: 0 = row-major C; 1 = q/k per-head [bh][i][c]; 2 = v per-head [bh][c][i]
// ---------------------------------------------------------------------------
__global__ __launch_bounds__(128) void gemm_bf16_kernel(
    const unsigned short* __restrict__ A, const unsigned short* __restrict__ BT,
    int M, int N, int K, const float* __restrict__ bias, int relu,
    float* __restrict__ Cf, unsigned short* __restrict__ Cb, int omode) {
  int m0   = blockIdx.x * 16;
  int wave = threadIdx.x >> 5;
  int lane = threadIdx.x & 31;
  int n0   = blockIdx.y * 64 + wave * 16;
  v8f acc = {};
#pragma unroll 2
  for (int k0 = 0; k0 < K; k0 += 32) {
    vu16x16 af = load_frag_a(A, K, m0, k0);
    vu16x16 bf = load_frag_b(BT, K, n0, k0);
    acc = wmma_bf16(af, bf, acc);
  }
  int n  = n0 + (lane & 15);
  int mb = (lane & 16) ? 8 : 0;
  float bv = bias ? bias[n] : 0.0f;
  int h = n & 7, c = n >> 3;          // column n = c*H + h (head split)
#pragma unroll
  for (int i = 0; i < 8; ++i) {
    int m = m0 + mb + i;
    float v = acc[i] + bv;
    if (relu) v = fmaxf(v, 0.0f);
    if (Cf) Cf[(long)m * N + n] = v;
    if (Cb) {
      long idx;
      if (omode == 0) {
        idx = (long)m * N + n;
      } else {
        int bb = m >> 9, ii = m & (NN - 1);
        if (omode == 1) idx = ((long)(bb * HH + h) * NN + ii) * CHD + c;
        else            idx = ((long)(bb * HH + h) * CHD + c) * NN + ii;
      }
      Cb[idx] = f2bf(v);
    }
  }
}

// ---------------------------------------------------------------------------
// Fused attention: scores (WMMA) + relpos bias -> softmax -> a16 (bf16)
// plus the 129 relative-position bins of o_pair written straight into cat.
// Block = 256 threads (8 waves) = one (b,h,16-row tile); scores in 32 KB LDS.
// q16/k16 are per-head [b*H+h][i][c] (ld = CH = 32) -> pure vector fragments.
// ---------------------------------------------------------------------------
__global__ __launch_bounds__(256) void attn_softmax_kernel(
    const unsigned short* __restrict__ q16, const unsigned short* __restrict__ k16,
    const float* __restrict__ Wb, unsigned short* __restrict__ a16,
    unsigned short* __restrict__ cat) {
  __shared__ float sc[16][NN];
  int blk = blockIdx.x;
  int it = blk & 31;
  int h  = (blk >> 5) & 7;
  int b  = blk >> 8;
  int i0 = it * 16;
  int wave = threadIdx.x >> 5;
  int lane = threadIdx.x & 31;

  const unsigned short* qh = q16 + (long)(b * HH + h) * NN * CHD;
  const unsigned short* kh = k16 + (long)(b * HH + h) * NN * CHD;

  // ---- phase 1: scores = qk * (W_L/sqrt(CH)) + W_L*bias  (K=CH=32: 1 WMMA)
  vu16x16 af = load_frag_a(qh, CHD, i0, 0);
  for (int jt = wave * 4; jt < wave * 4 + 4; ++jt) {
    int j0 = jt * 16;
    vu16x16 bf = load_frag_b(kh, CHD, j0, 0);   // B[c,j] = k[j,c] via k-head rows
    v8f acc = {};
    acc = wmma_bf16(af, bf, acc);
    int n  = lane & 15;
    int mb = (lane & 16) ? 8 : 0;
    int gj = j0 + n;
#pragma unroll
    for (int i = 0; i < 8; ++i) {
      int gi = i0 + mb + i;
      int d = gi - gj + KK;
      d = d < 0 ? 0 : (d > 2 * KK ? 2 * KK : d);
      float bv = Wb[d * HH + h];
      sc[mb + i][gj] = acc[i] * 0.125f + 0.70710678118654752f * bv;
    }
  }
  __syncthreads();

  // ---- phase 2: per-row softmax + o_pair bins; wave w -> rows w and w+8
#pragma unroll
  for (int rr = 0; rr < 2; ++rr) {
    int r  = wave + rr * 8;
    int gi = i0 + r;
    float xs[16];
    float mx = -1e30f;
#pragma unroll
    for (int u = 0; u < 16; ++u) {
      float x = sc[r][lane + u * 32];
      xs[u] = x;
      mx = fmaxf(mx, x);
    }
    for (int o = 16; o; o >>= 1) mx = fmaxf(mx, __shfl_xor(mx, o, 32));
    float tot = 0.f, s0 = 0.f, s128 = 0.f;
#pragma unroll
    for (int u = 0; u < 16; ++u) {
      int j = lane + u * 32;
      float e = __expf(xs[u] - mx);
      xs[u] = e;
      tot += e;
      if (j >= gi + KK) s0 += e;       // bin 0  : i-j+K <= 0
      if (j <= gi - KK) s128 += e;     // bin 2K : i-j+K >= 2K
    }
    for (int o = 16; o; o >>= 1) {
      tot  += __shfl_xor(tot,  o, 32);
      s0   += __shfl_xor(s0,   o, 32);
      s128 += __shfl_xor(s128, o, 32);
    }
    float rinv = 1.0f / tot;
    unsigned short* arow = a16 + ((long)(b * HH + h) * NN + gi) * NN;
#pragma unroll
    for (int u = 0; u < 16; ++u)
      arow[lane + u * 32] = f2bf(xs[u] * rinv);
    // o_pair bins: middle bins gather one element; ends use masked sums
    unsigned short* crow = cat + (long)(b * NN + gi) * CATP + h * BINS;
    for (int c = lane; c < BINS; c += 32) {
      float v;
      if (c == 0)         v = s0 * rinv;
      else if (c == 2*KK) v = s128 * rinv;
      else {
        int j = gi + KK - c;
        v = (j >= 0 && j < NN) ? __expf(sc[r][j] - mx) * rinv : 0.0f;
      }
      crow[c] = f2bf(v);
    }
  }
}

// ---------------------------------------------------------------------------
// o = a @ v, written into cat[..., H*BINS + h*CH + c].
// v16 is per-head TRANSPOSED [b*H+h][c][j] (ld = N) -> vector B fragments.
// Block = 64 threads (2 waves): one (b,h,16-row tile); wave -> 16-col c tile.
// ---------------------------------------------------------------------------
__global__ __launch_bounds__(64) void av_kernel(
    const unsigned short* __restrict__ a16, const unsigned short* __restrict__ v16,
    unsigned short* __restrict__ cat) {
  int blk = blockIdx.x;
  int it = blk & 31;
  int h  = (blk >> 5) & 7;
  int b  = blk >> 8;
  int i0 = it * 16;
  int wave = threadIdx.x >> 5;
  int lane = threadIdx.x & 31;
  const unsigned short* ah = a16 + (long)(b * HH + h) * NN * NN;
  const unsigned short* vh = v16 + (long)(b * HH + h) * CHD * NN;
  int c0 = wave * 16;
  v8f acc = {};
#pragma unroll 2
  for (int k0 = 0; k0 < NN; k0 += 32) {
    vu16x16 af = load_frag_a(ah, NN, i0, k0);
    vu16x16 bf = load_frag_b(vh, NN, c0, k0);   // BT[c][j] = v[j][c]
    acc = wmma_bf16(af, bf, acc);
  }
  int n  = c0 + (lane & 15);
  int mb = (lane & 16) ? 8 : 0;
#pragma unroll
  for (int i = 0; i < 8; ++i) {
    int gi = i0 + mb + i;
    cat[(long)(b * NN + gi) * CATP + HH * BINS + h * CHD + n] = f2bf(acc[i]);
  }
}

// ---------------------------------------------------------------------------
// out = LayerNorm(x + dlt)*g + be  -> f32 (+ optional bf16). One wave per row.
// ---------------------------------------------------------------------------
__global__ __launch_bounds__(256) void add_ln_kernel(
    const float* __restrict__ x, const float* __restrict__ dlt,
    const float* __restrict__ g, const float* __restrict__ be,
    float* __restrict__ outf, unsigned short* __restrict__ outb) {
  int row  = blockIdx.x * 8 + (threadIdx.x >> 5);
  int lane = threadIdx.x & 31;
  const float* xr = x + (long)row * CSD;
  const float* dr = dlt + (long)row * CSD;
  float v[8];
  float s = 0.f;
#pragma unroll
  for (int u = 0; u < 8; ++u) {
    v[u] = xr[lane + u * 32] + dr[lane + u * 32];
    s += v[u];
  }
  for (int o = 16; o; o >>= 1) s += __shfl_xor(s, o, 32);
  float mean = s * (1.0f / CSD);
  float var = 0.f;
#pragma unroll
  for (int u = 0; u < 8; ++u) { float d = v[u] - mean; var += d * d; }
  for (int o = 16; o; o >>= 1) var += __shfl_xor(var, o, 32);
  var *= (1.0f / CSD);
  float rs = rsqrtf(var + 1e-5f);
#pragma unroll
  for (int u = 0; u < 8; ++u) {
    int cc = lane + u * 32;
    float o = (v[u] - mean) * rs * g[cc] + be[cc];
    outf[(long)row * CSD + cc] = o;
    if (outb) outb[(long)row * CSD + cc] = f2bf(o);
  }
}

// ---------------------------------------------------------------------------
// out[b,:] = (sum_i s2[b,i,:]) @ Wout + N*bout
// ---------------------------------------------------------------------------
__global__ __launch_bounds__(256) void out_kernel(
    const float* __restrict__ s2, const float* __restrict__ Wout,
    const float* __restrict__ bout, float* __restrict__ out) {
  __shared__ float colsum[CSD];
  int b = blockIdx.x;
  int t = threadIdx.x;
  float s = 0.f;
  for (int i = 0; i < NN; ++i) s += s2[((long)b * NN + i) * CSD + t];
  colsum[t] = s;
  __syncthreads();
  if (t < OUTD) {
    float acc = (float)NN * bout[t];
    for (int c = 0; c < CSD; ++c) acc += colsum[c] * Wout[c * OUTD + t];
    out[b * OUTD + t] = acc;
  }
}

// ---------------------------------------------------------------------------
// Host side
// ---------------------------------------------------------------------------
extern "C" void kernel_launch(void* const* d_in, const int* in_sizes, int n_in,
                              void* d_out, int out_size, void* d_ws, size_t ws_size,
                              hipStream_t stream) {
  const float* s    = (const float*)d_in[0];
  const float* Wq   = (const float*)d_in[1];
  const float* Wk   = (const float*)d_in[2];
  const float* Wv   = (const float*)d_in[3];
  const float* Wb   = (const float*)d_in[4];
  const float* Wo   = (const float*)d_in[5];
  const float* bo   = (const float*)d_in[6];
  const float* g1   = (const float*)d_in[7];
  const float* be1  = (const float*)d_in[8];
  const float* W1   = (const float*)d_in[9];
  const float* b1   = (const float*)d_in[10];
  const float* W2   = (const float*)d_in[11];
  const float* b2   = (const float*)d_in[12];
  const float* g2   = (const float*)d_in[13];
  const float* be2  = (const float*)d_in[14];
  const float* Wout = (const float*)d_in[15];
  const float* bout = (const float*)d_in[16];
  float* out = (float*)d_out;

  char* ws = (char*)d_ws;
  size_t off = 0;
  auto alloc = [&](size_t bytes) -> void* {
    void* p = ws + off;
    off += (bytes + 255) & ~(size_t)255;
    return p;
  };
  // all B operands stored TRANSPOSED [N][K]
  unsigned short* s16  = (unsigned short*)alloc((size_t)ROWS * CSD * 2);
  unsigned short* WqT  = (unsigned short*)alloc((size_t)CSD * CSD * 2);
  unsigned short* WkT  = (unsigned short*)alloc((size_t)CSD * CSD * 2);
  unsigned short* WvT  = (unsigned short*)alloc((size_t)CSD * CSD * 2);
  unsigned short* W1T  = (unsigned short*)alloc((size_t)CTD * CSD * 2);  // [1024][256]
  unsigned short* W2T  = (unsigned short*)alloc((size_t)CSD * CTD * 2);  // [256][1024]
  unsigned short* WoT  = (unsigned short*)alloc((size_t)CSD * CATP * 2); // [256][1312]
  unsigned short* q16  = (unsigned short*)alloc((size_t)ROWS * CSD * 2); // [bh][i][c]
  unsigned short* k16  = (unsigned short*)alloc((size_t)ROWS * CSD * 2); // [bh][i][c]
  unsigned short* v16  = (unsigned short*)alloc((size_t)ROWS * CSD * 2); // [bh][c][j]
  unsigned short* a16  = (unsigned short*)alloc((size_t)BB * HH * NN * NN * 2);
  unsigned short* cat  = (unsigned short*)alloc((size_t)ROWS * CATP * 2);
  float* embd = (float*)alloc((size_t)ROWS * CSD * 4);
  float* s1f  = (float*)alloc((size_t)ROWS * CSD * 4);
  unsigned short* s1b = (unsigned short*)alloc((size_t)ROWS * CSD * 2);
  unsigned short* h1b = (unsigned short*)alloc((size_t)ROWS * CTD * 2);
  float* fff  = (float*)alloc((size_t)ROWS * CSD * 4);
  float* s2f  = (float*)alloc((size_t)ROWS * CSD * 4);

  // conversions: A operand plain, B operands transposed
  {
    long n = (long)ROWS * CSD;
    cvt_bf16_kernel<<<(int)((n + 255) / 256), 256, 0, stream>>>(s, s16, n);
  }
  auto cvtT = [&](const float* src, unsigned short* dst, int K, int N, int ldo) {
    long n = (long)K * N;
    cvt_bf16_t_kernel<<<(int)((n + 255) / 256), 256, 0, stream>>>(src, dst, K, N, ldo);
  };
  cvtT(Wq, WqT, CSD, CSD, CSD);
  cvtT(Wk, WkT, CSD, CSD, CSD);
  cvtT(Wv, WvT, CSD, CSD, CSD);
  cvtT(W1, W1T, CSD, CTD, CSD);
  cvtT(W2, W2T, CTD, CSD, CTD);
  cvtT(Wo, WoT, CATK, CSD, CATP);
  zero_pads_kernel<<<(ROWS * (CATP - CATK) + 255) / 256, 256, 0, stream>>>(cat, WoT);

  // q/k/v projections: 4096x256x256 bf16 WMMA GEMMs, head-split outputs
  dim3 gqkv(ROWS / 16, CSD / 64);
  gemm_bf16_kernel<<<gqkv, 128, 0, stream>>>(s16, WqT, ROWS, CSD, CSD, nullptr, 0, nullptr, q16, 1);
  gemm_bf16_kernel<<<gqkv, 128, 0, stream>>>(s16, WkT, ROWS, CSD, CSD, nullptr, 0, nullptr, k16, 1);
  gemm_bf16_kernel<<<gqkv, 128, 0, stream>>>(s16, WvT, ROWS, CSD, CSD, nullptr, 0, nullptr, v16, 2);

  // attention: scores + softmax + o_pair bins, then a@v
  attn_softmax_kernel<<<BB * HH * (NN / 16), 256, 0, stream>>>(q16, k16, Wb, a16, cat);
  av_kernel<<<BB * HH * (NN / 16), 64, 0, stream>>>(a16, v16, cat);

  // embd = cat @ Wo + bo  (K padded to 1312, WoT pre-transposed/padded)
  dim3 gemb(ROWS / 16, CSD / 64);
  gemm_bf16_kernel<<<gemb, 128, 0, stream>>>(cat, WoT, ROWS, CSD, CATP, bo, 0, embd, nullptr, 0);

  // s1 = LN(s + embd)
  add_ln_kernel<<<ROWS / 8, 256, 0, stream>>>(s, embd, g1, be1, s1f, s1b);

  // FFN
  dim3 gff1(ROWS / 16, CTD / 64);
  gemm_bf16_kernel<<<gff1, 128, 0, stream>>>(s1b, W1T, ROWS, CTD, CSD, b1, 1, nullptr, h1b, 0);
  dim3 gff2(ROWS / 16, CSD / 64);
  gemm_bf16_kernel<<<gff2, 128, 0, stream>>>(h1b, W2T, ROWS, CSD, CTD, b2, 0, fff, nullptr, 0);

  // s2 = LN(s1 + ff)
  add_ln_kernel<<<ROWS / 8, 256, 0, stream>>>(s1f, fff, g2, be2, s2f, nullptr);

  // final reduce + tiny GEMV
  out_kernel<<<BB, 256, 0, stream>>>(s2f, Wout, bout, out);

  (void)in_sizes; (void)n_in; (void)out_size; (void)ws_size;
}
